// SwinTransformerBlock_62680752718180
// MI455X (gfx1250) — matrix operand order repaired and checked by reading, not compiled, not run on hardware
//
#include <hip/hip_runtime.h>
#include <hip/hip_bf16.h>

// ---------------- Static problem config ----------------
#define HH 56
#define WW_ 56
#define CC 256
#define NHEAD 8
#define SS3 3
#define HD 32
#define NTOK 49           // tokens per window
#define NWIN 64           // windows per image
#define BATCH 32
#define TOKENS (BATCH * HH * WW_)       // 100352 (== BATCH*NWIN*NTOK)
#define HIDDEN 1024
#define SCALE 0.17677669529663687f      // HD^-0.5
#define LN_EPS 1e-5f

typedef __attribute__((ext_vector_type(16))) _Float16 v16h;
typedef __attribute__((ext_vector_type(8)))  float    v8f;

union FragU { uint4 u[2]; v16h v; };
union Chunk8 { uint4 u; _Float16 h[8]; };

// ---- CDNA5 async global->LDS copy (ASYNCcnt-tracked, no VGPR round trip) --
// LDS destination passed as 32-bit LDS byte address (low bits of the generic
// pointer to a __shared__ object), global source as 64-bit address pair.
__device__ __forceinline__ void async_copy_b128(unsigned int lds_off, const void* g) {
  asm volatile("global_load_async_to_lds_b128 %0, %1, off"
               :: "v"(lds_off), "v"((unsigned long long)(size_t)g)
               : "memory");
}
__device__ __forceinline__ void wait_async_le6() {
  asm volatile("s_wait_asynccnt 0x6" ::: "memory");
}
__device__ __forceinline__ void wait_async0() {
  asm volatile("s_wait_asynccnt 0x0" ::: "memory");
}

__device__ __forceinline__ float wave_sum(float v) {
#pragma unroll
  for (int o = 16; o > 0; o >>= 1) v += __shfl_xor(v, o, 32);
  return v;
}

// -------- f32 -> f16 weight conversion + fragment-major swizzle -----------
// Input: row-major [K][N]. Output: [N/16][K/32][lane 0..31][e 0..15] halves,
// where lane = (n&15) + 16*((k>>4)&1), e = k&15  (B-fragment layout).
__global__ void cvt_swizzle_b_kernel(const float* __restrict__ in,
                                     _Float16* __restrict__ out, int K, int N) {
  int i = blockIdx.x * 256 + threadIdx.x;
  if (i >= K * N) return;
  int k = i / N, n = i - k * N;
  int nt = n >> 4, kb = k >> 5;
  int lane = (n & 15) + ((k >> 4) & 1) * 16;
  int e = k & 15;
  out[(((size_t)nt * (K >> 5) + kb) * 32 + lane) * 16 + e] = (_Float16)in[i];
}

// ---------------- LN1 + cyclic shift + window partition -> f16 rows --------
__global__ __launch_bounds__(256) void ln_shift_window_kernel(
    const float* __restrict__ x, const float* __restrict__ g,
    const float* __restrict__ be, _Float16* __restrict__ out) {
  int wid  = blockIdx.x * 8 + (threadIdx.x >> 5);   // destination (windowed) row
  int lane = threadIdx.x & 31;
  int b = wid / 3136, rem = wid % 3136;
  int widx = rem / 49, inner = rem % 49;
  int hs = (widx >> 3) * 7 + inner / 7;
  int ws = (widx & 7) * 7 + inner % 7;
  int h = hs + SS3; if (h >= HH)  h -= HH;          // roll(-SS): src = dst+SS
  int w = ws + SS3; if (w >= WW_) w -= WW_;
  const float* src = x + ((size_t)b * 3136 + h * 56 + w) * CC;
  float4 v0 = *(const float4*)(src + lane * 8);
  float4 v1 = *(const float4*)(src + lane * 8 + 4);
  float s = v0.x + v0.y + v0.z + v0.w + v1.x + v1.y + v1.z + v1.w;
  float mu = wave_sum(s) * (1.f / CC);
  float q = (v0.x-mu)*(v0.x-mu) + (v0.y-mu)*(v0.y-mu) + (v0.z-mu)*(v0.z-mu) + (v0.w-mu)*(v0.w-mu)
          + (v1.x-mu)*(v1.x-mu) + (v1.y-mu)*(v1.y-mu) + (v1.z-mu)*(v1.z-mu) + (v1.w-mu)*(v1.w-mu);
  float rstd = rsqrtf(wave_sum(q) * (1.f / CC) + LN_EPS);
  float vv[8] = {v0.x, v0.y, v0.z, v0.w, v1.x, v1.y, v1.z, v1.w};
  _Float16* dst = out + (size_t)wid * CC + lane * 8;
#pragma unroll
  for (int i = 0; i < 8; ++i) {
    int c = lane * 8 + i;
    dst[i] = (_Float16)((vv[i] - mu) * rstd * g[c] + be[c]);
  }
}

// ---------------- plain LN (token order) -> f16 ----------------
__global__ __launch_bounds__(256) void ln_plain_kernel(
    const float* __restrict__ x, const float* __restrict__ g,
    const float* __restrict__ be, _Float16* __restrict__ out) {
  int tok  = blockIdx.x * 8 + (threadIdx.x >> 5);
  int lane = threadIdx.x & 31;
  const float* src = x + (size_t)tok * CC;
  float4 v0 = *(const float4*)(src + lane * 8);
  float4 v1 = *(const float4*)(src + lane * 8 + 4);
  float s = v0.x + v0.y + v0.z + v0.w + v1.x + v1.y + v1.z + v1.w;
  float mu = wave_sum(s) * (1.f / CC);
  float q = (v0.x-mu)*(v0.x-mu) + (v0.y-mu)*(v0.y-mu) + (v0.z-mu)*(v0.z-mu) + (v0.w-mu)*(v0.w-mu)
          + (v1.x-mu)*(v1.x-mu) + (v1.y-mu)*(v1.y-mu) + (v1.z-mu)*(v1.z-mu) + (v1.w-mu)*(v1.w-mu);
  float rstd = rsqrtf(wave_sum(q) * (1.f / CC) + LN_EPS);
  float vv[8] = {v0.x, v0.y, v0.z, v0.w, v1.x, v1.y, v1.z, v1.w};
  _Float16* dst = out + (size_t)tok * CC + lane * 8;
#pragma unroll
  for (int i = 0; i < 8; ++i) {
    int c = lane * 8 + i;
    dst[i] = (_Float16)((vv[i] - mu) * rstd * g[c] + be[c]);
  }
}

// ---------------- window reverse + un-shift + residual add (f32) ----------
__global__ __launch_bounds__(256) void unshift_add_kernel(
    const float* __restrict__ x, const float* __restrict__ o_win,
    float* __restrict__ xres) {
  int tok  = blockIdx.x * 8 + (threadIdx.x >> 5);   // b*3136 + h*56 + w
  int lane = threadIdx.x & 31;
  int b = tok / 3136, rem = tok % 3136;
  int h = rem / 56, w = rem % 56;
  int hs = h + (HH - SS3); if (hs >= HH)  hs -= HH; // (h - SS) mod H
  int ws = w + (WW_ - SS3); if (ws >= WW_) ws -= WW_;
  int row = (b * NWIN + (hs / 7) * 8 + (ws / 7)) * NTOK + (hs % 7) * 7 + (ws % 7);
  const float* xo = x + (size_t)tok * CC;
  const float* oo = o_win + (size_t)row * CC;
  float* r = xres + (size_t)tok * CC;
#pragma unroll
  for (int i = 0; i < 2; ++i) {
    int c = lane * 8 + i * 4;
    float4 a = *(const float4*)(xo + c);
    float4 o = *(const float4*)(oo + c);
    a.x += o.x; a.y += o.y; a.z += o.z; a.w += o.w;
    *(float4*)(r + c) = a;
  }
}

// ------------- WMMA GEMM : C = A(f16)·Bsw(f16, pre-swizzled) + bias -------
// Block: 128 thr (4 waves), tile 128(M) x 64(N). Wave w owns M-strips
// {w, w+4}; 8 WMMA per K-step per wave. Double-buffered LDS filled with
// CDNA5 async global->LDS copies overlapped with WMMA compute.
// EPI: 0 = store f16 ; 1 = GELU store f16 ; 2 = +resid store f32 ; 3 = f32
template <int EPI>
__global__ __launch_bounds__(128) void gemm_f16_wmma(
    const _Float16* __restrict__ A, const _Float16* __restrict__ Bsw,
    const float* __restrict__ bias, const float* __restrict__ resid,
    void* __restrict__ outp, int M, int N, int K) {
  __shared__ __align__(16) _Float16 As[2][128 * 32];   // [buf][strip8][lane32][e16]
  __shared__ __align__(16) _Float16 Bs[2][4 * 32 * 16];// [buf][nt4][lane32][e16]
  const int tid = threadIdx.x;
  const int wave = tid >> 5, lane = tid & 31;
  const int bm = blockIdx.y * 128, bn = blockIdx.x * 64;
  const int KB = K >> 5;
  v8f acc[2][4] = {};

  auto stage = [&](int kb, int buf) {
    // A tile (128x32): 512 chunks of 8 halves, swizzle is store-side free
#pragma unroll
    for (int i = 0; i < 4; ++i) {
      int c = tid + i * 128;
      int ra = c >> 2, ca = (c & 3) << 3;
      int s = ra >> 4, m = ra & 15;
      int hi = (ca >> 4) & 1, eo = ca & 8;
      unsigned int dst =
          (unsigned int)(size_t)&As[buf][((s * 32 + hi * 16 + m) << 4) + eo];
      async_copy_b128(dst, &A[(size_t)(bm + ra) * K + (kb << 5) + ca]);
    }
    // B tile: contiguous copy from pre-swizzled weights (L2 resident)
#pragma unroll
    for (int i = 0; i < 2; ++i) {
      int c = tid + i * 128;
      int t = c >> 6, j = c & 63;
      unsigned int dst = (unsigned int)(size_t)&Bs[buf][c * 8];
      async_copy_b128(dst,
          &Bsw[(((size_t)(bn >> 4) + t) * KB + kb) * 512 + j * 8]);
    }
  };

  stage(0, 0);
  for (int kb = 0; kb < KB; ++kb) {
    const int cur = kb & 1;
    if (kb + 1 < KB) {
      stage(kb + 1, cur ^ 1);   // prefetch next tile into the other buffer
      wait_async_le6();         // FIFO: <=6 outstanding => current tile done
    } else {
      wait_async0();
    }
    __syncthreads();
    FragU bf[4];
#pragma unroll
    for (int t = 0; t < 4; ++t) {
      const _Float16* bp = &Bs[cur][(t * 32 + lane) << 4];
      bf[t].u[0] = *(const uint4*)(bp);
      bf[t].u[1] = *(const uint4*)(bp + 8);
    }
#pragma unroll
    for (int ss = 0; ss < 2; ++ss) {
      FragU af;
      const _Float16* ap = &As[cur][((ss * 4 + wave) * 32 + lane) << 4];
      af.u[0] = *(const uint4*)(ap);
      af.u[1] = *(const uint4*)(ap + 8);
#pragma unroll
      for (int t = 0; t < 4; ++t)
        acc[ss][t] = __builtin_amdgcn_wmma_f32_16x16x32_f16(
            false, af.v, false, bf[t].v, (short)0, acc[ss][t], false, false);
    }
    __syncthreads();  // readers done before next iteration overwrites buf
  }
  int ncol = lane & 15, hic = lane >> 4;
#pragma unroll
  for (int ss = 0; ss < 2; ++ss) {
#pragma unroll
    for (int t = 0; t < 4; ++t) {
      int gn = bn + t * 16 + ncol;
      float bv = bias[gn];
#pragma unroll
      for (int v = 0; v < 8; ++v) {
        int gm = bm + (ss * 4 + wave) * 16 + v + 8 * hic;
        float xv = acc[ss][t][v] + bv;
        if (EPI == 1) xv = 0.5f * xv * (1.f + erff(xv * 0.7071067811865475f));
        if (EPI == 0 || EPI == 1)
          ((_Float16*)outp)[(size_t)gm * N + gn] = (_Float16)xv;
        else if (EPI == 2)
          ((float*)outp)[(size_t)gm * N + gn] = xv + resid[(size_t)gm * N + gn];
        else
          ((float*)outp)[(size_t)gm * N + gn] = xv;
      }
    }
  }
}

// ---------------- fused per-(window,head) attention ----------------
// Block = 128 thr (4 waves). blockIdx.x = bw*8 + head. All matrix tiles
// are kept in fragment-major LDS layout (b128 fragment loads). Q and K^T
// are staged with CDNA5 async global->LDS copies.
__global__ __launch_bounds__(128) void attn_kernel(
    const _Float16* __restrict__ qkv, const float* __restrict__ rpb,
    _Float16* __restrict__ outp) {
  __shared__ __align__(16) _Float16 Qsw[4 * 32 * 16];      // [strip][lane][e]
  __shared__ __align__(16) _Float16 KTsw[4 * 32 * 16];     // [nt][lane][e]
  __shared__ __align__(16) _Float16 Vsw[2 * 2 * 32 * 16];  // [ks][nt][lane][e]
  __shared__ __align__(16) _Float16 Psw[2 * 4 * 32 * 16];  // [ks][strip][lane][e]
  __shared__ float Ss[64 * 64];
  __shared__ int regS[64];
  const int tid = threadIdx.x;
  const int wave = tid >> 5, lane = tid & 31;
  const int head = blockIdx.x & 7;
  const int bw = blockIdx.x >> 3;
  const int win = bw & (NWIN - 1);
  const int wh = win >> 3, ww = win & 7;
  // zero all f16 tiles (padding rows/cols must be exact zero)
  {
    uint4 z = {0, 0, 0, 0};
    for (int i = tid; i < (2048 * 3 + 4096) / 8; i += 128)
      ((uint4*)Qsw)[i] = z;  // Qsw,KTsw,Vsw,Psw are laid out consecutively
  }
  if (tid < 64) {
    int r = tid;
    if (r < NTOK) {
      int hc = wh * 7 + r / 7, wc = ww * 7 + r % 7;   // img-mask coords
      int rh = (hc < 49) ? 0 : (hc < 53 ? 1 : 2);
      int rw = (wc < 49) ? 0 : (wc < 53 ? 1 : 2);
      regS[r] = rh * 3 + rw;
    } else {
      regS[r] = -1;
    }
  }
  __syncthreads();  // zero-fill complete before async writes land
  // stage q (A-layout, async), k^T (B-layout, async), v (B-layout, scatter)
  for (int c = tid; c < NTOK * 4; c += 128) {
    int row = c >> 2, col = (c & 3) << 3;
    size_t base = ((size_t)(bw * NTOK + row)) * 768 + head * HD + col;
    // Q: A-fragment layout, contiguous async copy
    {
      int s = row >> 4, m = row & 15;
      int hi = (col >> 4) & 1, eo = col & 8;
      unsigned int dst =
          (unsigned int)(size_t)&Qsw[((s * 32 + hi * 16 + m) << 4) + eo];
      async_copy_b128(dst, &qkv[base]);
    }
    // K^T: B-fragment layout; a K-row chunk is one contiguous lane span
    {
      int nt = row >> 4;
      int lnK = (row & 15) + (col & 16);
      int eo = col & 8;
      unsigned int dst =
          (unsigned int)(size_t)&KTsw[((nt * 32 + lnK) << 4) + eo];
      async_copy_b128(dst, &qkv[base + 256]);
    }
    // V: B-fragment layout for the P@V GEMM (k = token row), 8x b16 scatter
    {
      Chunk8 vv; vv.u = *(const uint4*)(&qkv[base + 512]);
      int ks = row >> 5, hi = (row >> 4) & 1, e = row & 15;
      int nt = col >> 4;
#pragma unroll
      for (int j = 0; j < 8; ++j) {
        int lnV = ((col + j) & 15) + 16 * hi;
        Vsw[(((ks * 2 + nt) * 32 + lnV) << 4) + e] = vv.h[j];
      }
    }
  }
  wait_async0();
  __syncthreads();
  // S = Q · K^T  (M=64 pad, N=64 pad, K=32)
  {
    FragU af;
    const _Float16* ap = &Qsw[(wave * 32 + lane) << 4];
    af.u[0] = *(const uint4*)(ap);
    af.u[1] = *(const uint4*)(ap + 8);
    int ncol = lane & 15, hic = lane >> 4;
#pragma unroll
    for (int t = 0; t < 4; ++t) {
      FragU bf;
      const _Float16* bp = &KTsw[(t * 32 + lane) << 4];
      bf.u[0] = *(const uint4*)(bp);
      bf.u[1] = *(const uint4*)(bp + 8);
      v8f c = {};
      c = __builtin_amdgcn_wmma_f32_16x16x32_f16(
          false, af.v, false, bf.v, (short)0, c, false, false);
#pragma unroll
      for (int v = 0; v < 8; ++v)
        Ss[(wave * 16 + v + 8 * hic) * 64 + t * 16 + ncol] = c[v];
    }
  }
  __syncthreads();
  // softmax over valid 49 columns; P written directly in A-fragment layout
  if (tid < NTOK) {
    int r = tid;
    int i1 = r / 7, j1 = r % 7, rg = regS[r];
    float mx = -1e30f;
    for (int m = 0; m < NTOK; ++m) {
      int i2 = m / 7, j2 = m % 7;
      float bias = rpb[((i1 - i2 + 6) * 13 + (j1 - j2 + 6)) * NHEAD + head];
      float msk = (regS[m] == rg) ? 0.f : -100.f;
      float lg = Ss[r * 64 + m] * SCALE + bias + msk;
      Ss[r * 64 + m] = lg;
      mx = fmaxf(mx, lg);
    }
    float sum = 0.f;
    for (int m = 0; m < NTOK; ++m) {
      float e = __expf(Ss[r * 64 + m] - mx);
      Ss[r * 64 + m] = e;
      sum += e;
    }
    float inv = 1.f / sum;
    int s = r >> 4, rm = r & 15;
    for (int m = 0; m < NTOK; ++m) {
      int ks = m >> 5, kk = m & 31;
      int hi = (kk >> 4) & 1;
      int e = kk & 15;
      Psw[(((ks * 4 + s) * 32 + rm + 16 * hi) << 4) + e] =
          (_Float16)(Ss[r * 64 + m] * inv);
    }
  }
  __syncthreads();
  // O = P · V  (K=64 in 2 steps, N=32 in 2 tiles)
  v8f acc[2] = {};
#pragma unroll
  for (int ks = 0; ks < 2; ++ks) {
    FragU af;
    const _Float16* ap = &Psw[((ks * 4 + wave) * 32 + lane) << 4];
    af.u[0] = *(const uint4*)(ap);
    af.u[1] = *(const uint4*)(ap + 8);
#pragma unroll
    for (int t = 0; t < 2; ++t) {
      FragU bf;
      const _Float16* bp = &Vsw[((ks * 2 + t) * 32 + lane) << 4];
      bf.u[0] = *(const uint4*)(bp);
      bf.u[1] = *(const uint4*)(bp + 8);
      acc[t] = __builtin_amdgcn_wmma_f32_16x16x32_f16(
          false, af.v, false, bf.v, (short)0, acc[t], false, false);
    }
  }
  int ncol = lane & 15, hic = lane >> 4;
#pragma unroll
  for (int t = 0; t < 2; ++t) {
#pragma unroll
    for (int v = 0; v < 8; ++v) {
      int m = wave * 16 + v + 8 * hic;
      if (m < NTOK)
        outp[((size_t)(bw * NTOK + m)) * CC + head * HD + t * 16 + ncol] =
            (_Float16)acc[t][v];
    }
  }
}

// ---------------- host-side orchestration ----------------
extern "C" void kernel_launch(void* const* d_in, const int* in_sizes, int n_in,
                              void* d_out, int out_size, void* d_ws, size_t ws_size,
                              hipStream_t stream) {
  const float* x       = (const float*)d_in[0];
  const float* gamma1  = (const float*)d_in[1];
  const float* beta1   = (const float*)d_in[2];
  const float* qkv_w   = (const float*)d_in[3];
  const float* qkv_b   = (const float*)d_in[4];
  const float* proj_w  = (const float*)d_in[5];
  const float* proj_b  = (const float*)d_in[6];
  const float* rpb     = (const float*)d_in[7];
  const float* gamma2  = (const float*)d_in[8];
  const float* beta2   = (const float*)d_in[9];
  const float* fc1_w   = (const float*)d_in[10];
  const float* fc1_b   = (const float*)d_in[11];
  const float* fc2_w   = (const float*)d_in[12];
  const float* fc2_b   = (const float*)d_in[13];
  float* out = (float*)d_out;

  // workspace layout
  char* p = (char*)d_ws;
  size_t off = 0;
  auto alloc = [&](size_t bytes) {
    void* r = p + off;
    off += (bytes + 255) & ~(size_t)255;
    return r;
  };
  _Float16* qkvW16 = (_Float16*)alloc((size_t)CC * 3 * CC * 2);       // swizzled 256x768
  _Float16* projW16 = (_Float16*)alloc((size_t)CC * CC * 2);          // swizzled 256x256
  _Float16* fc1W16 = (_Float16*)alloc((size_t)CC * HIDDEN * 2);       // swizzled 256x1024
  _Float16* fc2W16 = (_Float16*)alloc((size_t)HIDDEN * CC * 2);       // swizzled 1024x256
  _Float16* winA  = (_Float16*)alloc((size_t)TOKENS * CC * 2);        // LN1 windows / later LN2 out
  _Float16* qkvBuf = (_Float16*)alloc((size_t)TOKENS * HIDDEN * 2);   // qkv (768 cols) / later h1 (1024)
  _Float16* attnO = (_Float16*)alloc((size_t)TOKENS * CC * 2);
  float* oWin = (float*)alloc((size_t)TOKENS * CC * 4);
  float* xres = (float*)alloc((size_t)TOKENS * CC * 4);
  _Float16* ln2o = winA;  // alias: winA is dead after QKV GEMM

  // 0) weight conversion + fragment swizzle
  cvt_swizzle_b_kernel<<<(CC*3*CC + 255)/256, 256, 0, stream>>>(qkv_w, qkvW16, CC, 3*CC);
  cvt_swizzle_b_kernel<<<(CC*CC + 255)/256, 256, 0, stream>>>(proj_w, projW16, CC, CC);
  cvt_swizzle_b_kernel<<<(CC*HIDDEN + 255)/256, 256, 0, stream>>>(fc1_w, fc1W16, CC, HIDDEN);
  cvt_swizzle_b_kernel<<<(HIDDEN*CC + 255)/256, 256, 0, stream>>>(fc2_w, fc2W16, HIDDEN, CC);

  // 1) LN1 + shift + window partition
  ln_shift_window_kernel<<<TOKENS/8, 256, 0, stream>>>(x, gamma1, beta1, winA);

  // 2) QKV GEMM: [100352,256] x [256,768]
  gemm_f16_wmma<0><<<dim3(768/64, TOKENS/128), 128, 0, stream>>>(
      winA, qkvW16, qkv_b, nullptr, qkvBuf, TOKENS, 768, CC);

  // 3) fused window attention (B*NW*NH blocks)
  attn_kernel<<<BATCH * NWIN * NHEAD, 128, 0, stream>>>(qkvBuf, rpb, attnO);

  // 4) proj GEMM -> f32 window-ordered
  gemm_f16_wmma<3><<<dim3(CC/64, TOKENS/128), 128, 0, stream>>>(
      attnO, projW16, proj_b, nullptr, oWin, TOKENS, CC, CC);

  // 5) window reverse + un-shift + residual
  unshift_add_kernel<<<TOKENS/8, 256, 0, stream>>>(x, oWin, xres);

  // 6) LN2
  ln_plain_kernel<<<TOKENS/8, 256, 0, stream>>>(xres, gamma2, beta2, ln2o);

  // 7) fc1 GEMM + exact GELU
  gemm_f16_wmma<1><<<dim3(HIDDEN/64, TOKENS/128), 128, 0, stream>>>(
      ln2o, fc1W16, fc1_b, nullptr, qkvBuf, TOKENS, HIDDEN, CC);

  // 8) fc2 GEMM + residual -> d_out (f32)
  gemm_f16_wmma<2><<<dim3(CC/64, TOKENS/128), 128, 0, stream>>>(
      qkvBuf, fc2W16, fc2_b, xres, out, TOKENS, CC, HIDDEN);
}
